// CustomConv2d_32538672234916
// MI455X (gfx1250) — matrix operand order repaired and checked by reading, not compiled, not run on hardware
//
#include <hip/hip_runtime.h>
#include <hip/hip_bf16.h>

typedef __attribute__((ext_vector_type(16))) _Float16 v16h;
typedef __attribute__((ext_vector_type(8)))  _Float16 v8h;
typedef __attribute__((ext_vector_type(8)))  float    v8f;
typedef __attribute__((ext_vector_type(4)))  float    v4f;

// ---------------------------------------------------------------------------
// Problem: x[16,64,64,64] f32, w[128,64,3,3] f32, bias[128] f32 -> out[16,128,64,64]
// out = 576 - conv(exp(x), exp(w), stride1, pad1) + bias   (pad contributes exp(0)=1)
// Implicit GEMM: M = B*H*W = 65536, N = Cout = 128, K = 576, via
// v_wmma_f32_16x16x32_f16 (f32 accumulate), 18 K-steps.
// Wave tiling: each wave = 1 N-tile (16 couts) x 4 M-tiles (64 pixels), so a
// single global B-fragment load feeds 4 WMMAs and every B fragment is read
// exactly once per block.
// ---------------------------------------------------------------------------

#define CH_STRIDE 72   // 64 channels + pad (144B row stride spreads LDS banks)

// ===== Kernel 1: pack exp(weight) into WMMA B fragments (f16) ===============
// frag = kidx*8 + ntile, kidx = pos*2 + kstep (pos = kh*3+kw, kstep = ci-half).
// Per-fragment layout follows the ISA 16-bit B 32x16 register order (7.12.4
// pattern): lane l holds N = l&15; lanes 0-15 hold K=0..7,16..23, lanes 16-31 hold
// K=8..15,24..31 -> 16 contiguous f16 per lane in memory.
__global__ __launch_bounds__(256) void pack_expw_kernel(
    const float* __restrict__ w, _Float16* __restrict__ Bw)
{
    const int tid = blockIdx.x * 256 + threadIdx.x;
    if (tid >= 144 * 512) return;
    const int frag  = tid >> 9;        // 0..143
    const int r     = tid & 511;
    const int lane  = r >> 4;          // 0..31
    const int e     = r & 15;          // f16 element within lane's v16h
    const int ntile = frag & 7;
    const int ks    = (frag >> 3) & 1;
    const int pos   = frag >> 4;       // 0..8  (kh*3+kw)
    const int nloc  = lane & 15;
    const int half  = lane >> 4;
    const int ci    = ks * 32 + half * 8 + (e & 7) + (e >> 3) * 16;   // channel = K within this pos
    const int cout  = (ntile << 4) + nloc;
    // weight layout [Cout][Cin][3][3]
    Bw[tid] = (_Float16)__expf(w[(cout * 64 + ci) * 9 + pos]);
}

// ===== A-fragment loader (16x32 f16 from LDS) ===============================
// Lane l holds M = l&15; V0-3 = K half*8+0..7, V4-7 = K 16+half*8+0..7.
__device__ __forceinline__ v16h load_a_frag(
    const _Float16 (*sA)[66][CH_STRIDE], int kidx, int col, int half)
{
    const int pos = kidx >> 1, ks = kidx & 1;   // constants under full unroll
    const int kh = pos / 3, kw = pos % 3;
    const _Float16* ap = &sA[kh][col + kw][ks * 32 + half * 8];
    const v8h lo = *(const v8h*)ap;             // ds_load_b128
    const v8h hi = *(const v8h*)(ap + 16);      // ds_load_b128
    return __builtin_shufflevector(lo, hi,
        0, 1, 2, 3, 4, 5, 6, 7, 8, 9, 10, 11, 12, 13, 14, 15);
}

// ===== Kernel 2: main implicit-GEMM conv ====================================
__global__ __launch_bounds__(256) void conv_exp_wmma_kernel(
    const float* __restrict__ x, const float* __restrict__ bias,
    const _Float16* __restrict__ Bw, float* __restrict__ out)
{
    // LDS: exp(x) for rows h-1..h+1, cols -1..64 (halo = exp(0) = 1.0), f16,
    // channel-contiguous: sA[row][col+1][ci]
    __shared__ _Float16 sA[3][66][CH_STRIDE];

    const int bh  = blockIdx.x;        // 0..1023
    const int b   = bh >> 6;
    const int h   = bh & 63;
    const int tid = threadIdx.x;

    // ---- warm the packed-B working set (147 KB, reused by all blocks) into
    //      near caches while we stage A; 1152 cachelines of 128 B.
    for (int line = tid; line < 1152; line += 256)
        __builtin_prefetch(Bw + line * 64, 0, 3);

    // ---- stage exp(x) into LDS (coalesced: consecutive tid -> consecutive col)
    for (int idx = tid; idx < 3 * 64 * 64; idx += 256) {
        const int row = idx >> 12;           // 0..2
        const int ci  = (idx >> 6) & 63;
        const int col = idx & 63;
        const int gh  = h + row - 1;
        float v = 1.0f;                      // exp(0) for zero-padded rows
        if (gh >= 0 && gh < 64)
            v = __expf(x[((b * 64 + ci) * 64 + gh) * 64 + col]);
        sA[row][col + 1][ci] = (_Float16)v;
    }
    // halo columns (image col -1 and 64) -> exp(0) = 1
    for (int idx = tid; idx < 3 * 2 * 64; idx += 256) {
        const int row  = idx >> 7;
        const int side = (idx >> 6) & 1;
        const int ci   = idx & 63;
        sA[row][side * 65][ci] = (_Float16)1.0f;
    }
    __syncthreads();

    // ---- wave tiling: 8 waves; wave w = N-tile w (couts w*16..w*16+15),
    //      each wave covers all 4 M-tiles (pixels 0..63 of this row).
    const int wave  = tid >> 5;              // = ntile
    const int lane  = tid & 31;
    const int m     = lane & 15;             // A-fragment M index / C N index
    const int half  = lane >> 4;

    // Per-wave B stream: frag = kidx*8 + wave, 512 f16 per frag.
    const _Float16* Bstream = Bw + wave * 512 + lane * 16;

    v8f acc[4] = {};                         // 4 M-tiles x (16x16 f32)

    // ---- prologue: B fragment for K-step 0
    v16h b_cur = *(const v16h*)Bstream;

    // ---- main loop: 18 K-steps; one B load feeds 4 WMMAs; B double-buffered
    #pragma unroll
    for (int k = 0; k < 18; ++k) {
        v16h b_nxt;
        if (k < 17)                          // compile-time under full unroll
            b_nxt = *(const v16h*)(Bstream + (k + 1) * 4096);
        // A fragments for the 4 M-tiles (LDS, low latency)
        v16h a[4];
        #pragma unroll
        for (int mt = 0; mt < 4; ++mt)
            a[mt] = load_a_frag(sA, k, mt * 16 + m, half);
        #pragma unroll
        for (int mt = 0; mt < 4; ++mt)
            acc[mt] = __builtin_amdgcn_wmma_f32_16x16x32_f16(
                false, a[mt], false, b_cur, (short)0, acc[mt], false, false);
        if (k < 17) b_cur = b_nxt;
    }

    // ---- epilogue: out = 576 - acc + bias
    // C layout: lane l -> N = l&15 (cout), M rows = half*8 + r (pixel w)
    const float KC = 576.0f;
    const int   o  = (wave << 4) + m;
    const float bo = bias[o];
    float* orow = out + (((b * 128 + o) * 64 + h) << 6) + half * 8;
    #pragma unroll
    for (int mt = 0; mt < 4; ++mt) {
        v4f r0, r1;
        #pragma unroll
        for (int i = 0; i < 4; ++i) {
            r0[i] = KC - acc[mt][i]     + bo;
            r1[i] = KC - acc[mt][i + 4] + bo;
        }
        float* op = orow + mt * 16;
        *(v4f*)op       = r0;   // 8 consecutive w per lane -> 2x b128 stores
        *(v4f*)(op + 4) = r1;
    }
}

// ===== host-side launcher ===================================================
extern "C" void kernel_launch(void* const* d_in, const int* in_sizes, int n_in,
                              void* d_out, int out_size, void* d_ws, size_t ws_size,
                              hipStream_t stream) {
    const float* x    = (const float*)d_in[0];   // [16,64,64,64]
    const float* w    = (const float*)d_in[1];   // [128,64,3,3]
    const float* bias = (const float*)d_in[2];   // [128]
    float*       out  = (float*)d_out;           // [16,128,64,64]
    _Float16*    Bw   = (_Float16*)d_ws;         // 144*512 f16 = 144 KiB packed B

    // Pack exp(weight) into WMMA B fragments (144*512 elements).
    pack_expw_kernel<<<288, 256, 0, stream>>>(w, Bw);
    // One block per (batch, output row): 16*64 = 1024 blocks.
    conv_exp_wmma_kernel<<<1024, 256, 0, stream>>>(x, bias, Bw, out);
}
